// VQVIB_37039797961386
// MI455X (gfx1250) — matrix operand order repaired and checked by reading, not compiled, not run on hardware
//
#include <hip/hip_runtime.h>
#include <hip/hip_bf16.h>
#include <math.h>

typedef __attribute__((ext_vector_type(16))) __bf16 v16bf;
typedef __attribute__((ext_vector_type(8)))  float  v8f;
typedef int vi4 __attribute__((vector_size(16)));   // matches builtin's V4i
typedef unsigned short u16;
typedef unsigned int   u32;

#define AS1 __attribute__((address_space(1)))
#define AS3 __attribute__((address_space(3)))

#define NB  32768   // batch
#define NIN 1024
#define NH  64
#define ND  512
#define NP  2048
#define LDP 520     // padded proto-chunk row stride (u16) to avoid LDS bank conflicts
#define H0P 72      // padded h0 row stride

#if __has_builtin(__builtin_amdgcn_global_load_async_to_lds_b128) && __has_builtin(__builtin_amdgcn_s_wait_asynccnt)
#define USE_ASYNC 1
#else
#define USE_ASYNC 0
#endif

__device__ __forceinline__ u16 f2bf(float f) {
    union { __bf16 b; u16 u; } x;
    x.b = (__bf16)f;              // hardware cvt path
    return x.u;
}

__device__ __forceinline__ v8f vzero8() {
    v8f z = {0.f,0.f,0.f,0.f,0.f,0.f,0.f,0.f};
    return z;
}

__device__ __forceinline__ v8f wmma_bf16(v16bf a, v16bf b, v8f c) {
    return __builtin_amdgcn_wmma_f32_16x16x32_bf16(false, a, false, b, (short)0, c, false, false);
}

// A fragment (16x32 bf16); `row` = base of this lane's M row.
// lanes 0-15: K = kb..kb+7, kb+16..kb+23 ; lanes 16-31: +8
__device__ __forceinline__ v16bf frag_a_bf16(const u16* row, int kb) {
    int hl = (threadIdx.x & 31) >> 4;
    union { v16bf v; uint4 q[2]; } u;
    u.q[0] = *(const uint4*)(row + kb + 8*hl);
    u.q[1] = *(const uint4*)(row + kb + 16 + 8*hl);
    return u.v;
}

// A fragment from row-major f32 (convert with hw cast)
__device__ __forceinline__ v16bf frag_a_f32(const float* row, int kb) {
    int hl = (threadIdx.x & 31) >> 4;
    union { v16bf v; u16 e[16]; } u;
    const float* p0 = row + kb + 8*hl;
    const float* p1 = row + kb + 16 + 8*hl;
#pragma unroll
    for (int j = 0; j < 8; ++j) { u.e[j] = f2bf(p0[j]); u.e[8+j] = f2bf(p1[j]); }
    return u.v;
}

// B fragment (32x16, B[k][n] = S[n][k]); `srow` = base of this lane's N row of S.
// lanes 0-15: K = kb..kb+7, kb+16..kb+23 ; lanes 16-31: +8
__device__ __forceinline__ v16bf frag_b_bf16(const u16* srow, int kb) {
    int hl = (threadIdx.x & 31) >> 4;
    union { v16bf v; uint4 q[2]; } u;
    u.q[0] = *(const uint4*)(srow + kb + 8*hl);
    u.q[1] = *(const uint4*)(srow + kb + 16 + 8*hl);
    return u.v;
}

// stage one 32-row proto chunk (bf16, padded rows) into LDS
__device__ __forceinline__ void stage_chunk(const u16* __restrict__ g, u16* pch) {
    for (int i = threadIdx.x; i < 32 * 64; i += 256) {     // 32 rows x 64 uint4
        int r = i >> 6, cq = i & 63;
#if USE_ASYNC
        __builtin_amdgcn_global_load_async_to_lds_b128(
            (AS1 vi4*)(u16*)(g + (size_t)r * ND + cq * 8),
            (AS3 vi4*)(pch + r * LDP + cq * 8),
            0, 0);
#else
        *(uint4*)(pch + r * LDP + cq * 8) = *(const uint4*)(g + (size_t)r * ND + cq * 8);
#endif
    }
#if USE_ASYNC
    __builtin_amdgcn_s_wait_asynccnt(0);
#endif
}

// ---------------- K0: zero accumulators (graph-replay safe) ----------------
__global__ void k0_init(float* softacc) {
    for (int i = threadIdx.x; i < NP + 2; i += 256) softacc[i] = 0.f;
}

// ---------------- K0b: protos -> bf16, proto norms ----------------
__global__ __launch_bounds__(256) void k0_protos(const float* __restrict__ protos,
                                                 u16* __restrict__ pb, float* __restrict__ pnorm) {
    int gt = blockIdx.x * 256 + threadIdx.x;
    for (int i = gt; i < NP * ND; i += 64 * 256) pb[i] = f2bf(protos[i]);
    if (gt < NP) {
        const float* r = protos + (size_t)gt * ND;
        float s = 0.f;
        for (int k = 0; k < ND; ++k) { float v = r[k]; s += v * v; }
        pnorm[gt] = s;
    }
}

// ---------------- K0c: weights -> bf16 (once, not per-block) ----------------
__global__ __launch_bounds__(256) void k0_weights(const float* __restrict__ W0, const float* __restrict__ W1,
                                                  const float* __restrict__ Wmu, const float* __restrict__ Wvar,
                                                  u16* __restrict__ w0b, u16* __restrict__ w1b,
                                                  u16* __restrict__ wmub, u16* __restrict__ wvarb) {
    int gt = blockIdx.x * 256 + threadIdx.x;
    int stride = gridDim.x * 256;
    for (int i = gt; i < NH * NIN; i += stride) w0b[i] = f2bf(W0[i]);
    for (int i = gt; i < ND * NH;  i += stride) w1b[i] = f2bf(W1[i]);
    for (int i = gt; i < ND * ND;  i += stride) wmub[i]  = f2bf(Wmu[i]);
    for (int i = gt; i < ND * ND;  i += stride) wvarb[i] = f2bf(Wvar[i]);
}

// ---------------- K1: fused encoder  h1 = relu(relu(x W0^T + b0) W1^T + b1) ----------------
__global__ __launch_bounds__(256) void k1_encoder(const float* __restrict__ x,
                                                  const u16* __restrict__ w0b, const float* __restrict__ b0,
                                                  const u16* __restrict__ w1b, const float* __restrict__ b1,
                                                  u16* __restrict__ h1b) {
    __shared__ __align__(16) u16 h0s[128 * H0P];
    const int lane = threadIdx.x & 31, wave = threadIdx.x >> 5;
    const int m = lane & 15, hl = lane >> 4;
    const int m0 = wave * 16;
    const int browg = blockIdx.x * 128;

    // phase 1: h0 (16 x 64 per wave), K = 1024
    v8f c[4];
#pragma unroll
    for (int nt = 0; nt < 4; ++nt) c[nt] = vzero8();
    const float* xrow = x + (size_t)(browg + m0 + m) * NIN;
    for (int kc = 0; kc < NIN; kc += 32) {
        v16bf a = frag_a_f32(xrow, kc);
#pragma unroll
        for (int nt = 0; nt < 4; ++nt)
            c[nt] = wmma_bf16(a, frag_b_bf16(w0b + (size_t)(nt * 16 + m) * NIN, kc), c[nt]);
    }
#pragma unroll
    for (int nt = 0; nt < 4; ++nt)
#pragma unroll
        for (int r = 0; r < 8; ++r) {
            int rl = m0 + 8 * hl + r, col = nt * 16 + m;
            float v = c[nt][r] + b0[col];
            h0s[rl * H0P + col] = f2bf(fmaxf(v, 0.f));
        }
    __syncthreads();

    // phase 2: h1 (16 x 512 per wave), K = 64
    const u16* arow = h0s + (m0 + m) * H0P;
    v16bf a2[2];
    a2[0] = frag_a_bf16(arow, 0);
    a2[1] = frag_a_bf16(arow, 32);
    for (int nc = 0; nc < ND; nc += 64) {
        v8f d[4];
#pragma unroll
        for (int nt = 0; nt < 4; ++nt) d[nt] = vzero8();
#pragma unroll
        for (int kc = 0; kc < 2; ++kc)
#pragma unroll
            for (int nt = 0; nt < 4; ++nt)
                d[nt] = wmma_bf16(a2[kc], frag_b_bf16(w1b + (size_t)(nc + nt * 16 + m) * NH, kc * 32), d[nt]);
#pragma unroll
        for (int nt = 0; nt < 4; ++nt)
#pragma unroll
            for (int r = 0; r < 8; ++r) {
                int grow = browg + m0 + 8 * hl + r, col = nc + nt * 16 + m;
                float v = d[nt][r] + b1[col];
                h1b[(size_t)grow * ND + col] = f2bf(fmaxf(v, 0.f));
            }
    }
}

// ---------------- K2: mu/logvar dual GEMM + reparameterize + KL partial ----------------
__global__ __launch_bounds__(256) void k2_latents(const u16* __restrict__ h1b,
                                                  const u16* __restrict__ wmub, const float* __restrict__ bmu,
                                                  const u16* __restrict__ wvarb, const float* __restrict__ bvar,
                                                  const float* __restrict__ eps,
                                                  float* __restrict__ latf, u16* __restrict__ latb,
                                                  float* __restrict__ kl_accum) {
    const int lane = threadIdx.x & 31, wave = threadIdx.x >> 5;
    const int m = lane & 15, hl = lane >> 4;
    const int m0 = wave * 16;
    const int browg = blockIdx.x * 128;
    const int ncol0 = blockIdx.y * 128;

    v8f cmu[8], cvar[8];
#pragma unroll
    for (int nt = 0; nt < 8; ++nt) { cmu[nt] = vzero8(); cvar[nt] = vzero8(); }

    const u16* arow = h1b + (size_t)(browg + m0 + m) * ND;
    for (int kc = 0; kc < ND; kc += 32) {
        v16bf a = frag_a_bf16(arow, kc);
#pragma unroll
        for (int nt = 0; nt < 8; ++nt) {
            int n = ncol0 + nt * 16 + m;
            cmu[nt]  = wmma_bf16(a, frag_b_bf16(wmub  + (size_t)n * ND, kc), cmu[nt]);
            cvar[nt] = wmma_bf16(a, frag_b_bf16(wvarb + (size_t)n * ND, kc), cvar[nt]);
        }
    }
    float kl = 0.f;
#pragma unroll
    for (int nt = 0; nt < 8; ++nt)
#pragma unroll
        for (int r = 0; r < 8; ++r) {
            int grow = browg + m0 + 8 * hl + r;
            int col = ncol0 + nt * 16 + m;
            float mu = cmu[nt][r] + bmu[col];
            float lv = cvar[nt][r] + bvar[col];
            float el = __expf(lv);
            float lat = mu + eps[(size_t)grow * ND + col] * __expf(0.5f * lv);
            size_t o = (size_t)grow * ND + col;
            latf[o] = lat;
            latb[o] = f2bf(lat);
            kl += 1.0f + lv - mu * mu - el;
        }
    __shared__ float red[256];
    red[threadIdx.x] = kl;
    __syncthreads();
    for (int s = 128; s > 0; s >>= 1) {
        if (threadIdx.x < s) red[threadIdx.x] += red[threadIdx.x + s];
        __syncthreads();
    }
    if (threadIdx.x == 0) atomicAdd(kl_accum, red[0]);
}

// ---------------- K3a: streaming scores, online logsumexp + argmax ----------------
__global__ __launch_bounds__(256) void k3a_lse(const u16* __restrict__ latb, const u16* __restrict__ pb,
                                               const float* __restrict__ pnorm,
                                               float* __restrict__ lse, int* __restrict__ idx) {
    __shared__ __align__(16) u16 pch[32 * LDP];
    const int lane = threadIdx.x & 31, wave = threadIdx.x >> 5;
    const int m = lane & 15, hl = lane >> 4;
    const int m0 = wave * 16;
    const int browg = blockIdx.x * 128;

    const u16* arow = latb + (size_t)(browg + m0 + m) * ND;
    v16bf a[16];
#pragma unroll
    for (int kc = 0; kc < 16; ++kc) a[kc] = frag_a_bf16(arow, kc * 32);

    float M[8], S[8], BV[8]; int BI[8];
#pragma unroll
    for (int r = 0; r < 8; ++r) { M[r] = -1e30f; S[r] = 0.f; BV[r] = -1e30f; BI[r] = 0; }

    for (int pc = 0; pc < NP; pc += 32) {
        __syncthreads();
        stage_chunk(pb + (size_t)pc * ND, pch);
        __syncthreads();
        v8f c[2];
        c[0] = vzero8(); c[1] = vzero8();
        for (int kc = 0; kc < 16; ++kc) {
#pragma unroll
            for (int nt = 0; nt < 2; ++nt)
                c[nt] = wmma_bf16(a[kc], frag_b_bf16(pch + (nt * 16 + m) * LDP, kc * 32), c[nt]);
        }
#pragma unroll
        for (int nt = 0; nt < 2; ++nt) {
            int p = pc + nt * 16 + m;
            float pn = pnorm[p];
#pragma unroll
            for (int r = 0; r < 8; ++r) {
                float s = 2.0f * c[nt][r] - pn;   // argmax(s)==argmin(dist); softmax(-d)==softmax(s)
                if (s > BV[r]) { BV[r] = s; BI[r] = p; }
                if (s > M[r]) { S[r] = S[r] * __expf(M[r] - s) + 1.0f; M[r] = s; }
                else          { S[r] += __expf(s - M[r]); }
            }
        }
    }
#pragma unroll
    for (int d = 1; d < 16; d <<= 1) {
#pragma unroll
        for (int r = 0; r < 8; ++r) {
            float Mo = __shfl_xor(M[r], d, 32);
            float So = __shfl_xor(S[r], d, 32);
            float nm = fmaxf(M[r], Mo);
            S[r] = S[r] * __expf(M[r] - nm) + So * __expf(Mo - nm);
            M[r] = nm;
            float bvo = __shfl_xor(BV[r], d, 32);
            int   bio = __shfl_xor(BI[r], d, 32);
            if (bvo > BV[r] || (bvo == BV[r] && bio < BI[r])) { BV[r] = bvo; BI[r] = bio; }
        }
    }
    if ((lane & 15) == 0) {
#pragma unroll
        for (int r = 0; r < 8; ++r) {
            int grow = browg + m0 + 8 * hl + r;
            lse[grow] = M[r] + __logf(S[r]);
            idx[grow] = BI[r];
        }
    }
}

// ---------------- K3b: probabilities, soft_dist sums, quantized gather, diff loss ----------------
__global__ __launch_bounds__(256) void k3b_vq(const u16* __restrict__ latb, const float* __restrict__ latf,
                                              const u16* __restrict__ pb, const float* __restrict__ pnorm,
                                              const float* __restrict__ protos,
                                              const float* __restrict__ lse, const int* __restrict__ idx,
                                              float* __restrict__ softacc, float* __restrict__ outq,
                                              float* __restrict__ diff_accum) {
    __shared__ __align__(16) u16 pch[32 * LDP];
    __shared__ float soft_lds[NP];
    for (int i = threadIdx.x; i < NP; i += 256) soft_lds[i] = 0.f;

    const int lane = threadIdx.x & 31, wave = threadIdx.x >> 5;
    const int m = lane & 15, hl = lane >> 4;
    const int m0 = wave * 16;
    const int browg = blockIdx.x * 128;

    const u16* arow = latb + (size_t)(browg + m0 + m) * ND;
    v16bf a[16];
#pragma unroll
    for (int kc = 0; kc < 16; ++kc) a[kc] = frag_a_bf16(arow, kc * 32);

    float L[8];
#pragma unroll
    for (int r = 0; r < 8; ++r) L[r] = lse[browg + m0 + 8 * hl + r];

    for (int pc = 0; pc < NP; pc += 32) {
        __syncthreads();
        stage_chunk(pb + (size_t)pc * ND, pch);
        __syncthreads();
        v8f c[2];
        c[0] = vzero8(); c[1] = vzero8();
        for (int kc = 0; kc < 16; ++kc) {
#pragma unroll
            for (int nt = 0; nt < 2; ++nt)
                c[nt] = wmma_bf16(a[kc], frag_b_bf16(pch + (nt * 16 + m) * LDP, kc * 32), c[nt]);
        }
#pragma unroll
        for (int nt = 0; nt < 2; ++nt) {
            int p = pc + nt * 16 + m;
            float pn = pnorm[p];
            float cs = 0.f;
#pragma unroll
            for (int r = 0; r < 8; ++r)
                cs += __expf(2.0f * c[nt][r] - pn - L[r]);
            cs += __shfl_xor(cs, 16, 32);            // combine row halves
            if (lane < 16) atomicAdd(&soft_lds[p], cs);
        }
    }

    // quantized output + (quantized - latents)^2
    float dsum = 0.f;
    for (int r16 = 0; r16 < 16; ++r16) {
        int grow = browg + m0 + r16;
        int qi = idx[grow];
        const float* qrow = protos + (size_t)qi * ND;
        const float* lrow = latf + (size_t)grow * ND;
        float* orow = outq + (size_t)grow * ND;
        for (int j = lane; j < ND; j += 32) {
            float q = qrow[j], l = lrow[j];
            orow[j] = q;
            float dd = q - l;
            dsum += dd * dd;
        }
    }
#pragma unroll
    for (int d = 16; d >= 1; d >>= 1) dsum += __shfl_xor(dsum, d, 32);
    if (lane == 0) atomicAdd(diff_accum, dsum);

    __syncthreads();
    for (int i = threadIdx.x; i < NP; i += 256) atomicAdd(&softacc[i], soft_lds[i]);
}

// ---------------- K4: finalize scalars ----------------
__global__ __launch_bounds__(256) void k4_final(const float* __restrict__ softacc,
                                                const float* __restrict__ kl_sum,
                                                const float* __restrict__ diff_sum,
                                                float* __restrict__ out) {
    __shared__ float red[256];
    __shared__ float Tsh;
    float t = 0.f;
    for (int i = threadIdx.x; i < NP; i += 256) t += softacc[i] / (float)NB + 1e-6f;
    red[threadIdx.x] = t; __syncthreads();
    for (int s = 128; s > 0; s >>= 1) {
        if (threadIdx.x < s) red[threadIdx.x] += red[threadIdx.x + s];
        __syncthreads();
    }
    if (threadIdx.x == 0) Tsh = red[0];
    __syncthreads();
    float T = Tsh;
    float e = 0.f;
    for (int i = threadIdx.x; i < NP; i += 256) {
        float v = (softacc[i] / (float)NB + 1e-6f) / T;
        e += -v * __logf(v);
    }
    red[threadIdx.x] = e; __syncthreads();
    for (int s = 128; s > 0; s >>= 1) {
        if (threadIdx.x < s) red[threadIdx.x] += red[threadIdx.x + s];
        __syncthreads();
    }
    if (threadIdx.x == 0) {
        float ent = red[0];
        float div = -0.5f * kl_sum[0] / (float)NB;
        float mse = diff_sum[0] / ((float)NB * (float)ND);
        float vq = 1.25f * mse + 0.1f * ent;   // commitment*0.25 + embedding (equal fwd values)
        out[(size_t)NB * ND]     = 0.01f * div + vq;
        out[(size_t)NB * ND + 1] = div;
    }
}

extern "C" void kernel_launch(void* const* d_in, const int* in_sizes, int n_in,
                              void* d_out, int out_size, void* d_ws, size_t ws_size,
                              hipStream_t stream) {
    const float* x     = (const float*)d_in[0];
    const float* W0    = (const float*)d_in[1];
    const float* b0    = (const float*)d_in[2];
    const float* W1    = (const float*)d_in[3];
    const float* b1    = (const float*)d_in[4];
    const float* Wmu   = (const float*)d_in[5];
    const float* bmu   = (const float*)d_in[6];
    const float* Wvar  = (const float*)d_in[7];
    const float* bvar  = (const float*)d_in[8];
    const float* protos= (const float*)d_in[9];
    const float* eps   = (const float*)d_in[10];

    char* ws = (char*)d_ws;
    u16*   h1b     = (u16*)  (ws + 0);                 //  32 MB  [B, 512] bf16
    float* latf    = (float*)(ws + 33554432);          //  64 MB  [B, 512] f32
    u16*   latb    = (u16*)  (ws + 100663296);         //  32 MB  [B, 512] bf16
    u16*   pb      = (u16*)  (ws + 134217728);         //   2 MB  [P, 512] bf16
    float* pnorm   = (float*)(ws + 136314880);         //  8 KB
    float* lse     = (float*)(ws + 136323072);         // 128 KB
    int*   idx     = (int*)  (ws + 136454144);         // 128 KB
    float* softacc = (float*)(ws + 136585216);         //  8 KB + 2 scalars
    float* kl_sum  = softacc + NP;
    float* diff_sum= softacc + NP + 1;
    u16*   w0b     = (u16*)  (ws + 136593664);         // 128 KB
    u16*   w1b     = (u16*)  (ws + 136724736);         //  64 KB
    u16*   wmub    = (u16*)  (ws + 136790272);         // 512 KB
    u16*   wvarb   = (u16*)  (ws + 137314560);         // 512 KB
    float* outq    = (float*)d_out;

    k0_init   <<<1, 256, 0, stream>>>(softacc);
    k0_protos <<<64, 256, 0, stream>>>(protos, pb, pnorm);
    k0_weights<<<64, 256, 0, stream>>>(W0, W1, Wmu, Wvar, w0b, w1b, wmub, wvarb);
    k1_encoder<<<NB / 128, 256, 0, stream>>>(x, w0b, b0, w1b, b1, h1b);
    k2_latents<<<dim3(NB / 128, ND / 128), 256, 0, stream>>>(h1b, wmub, bmu, wvarb, bvar, eps,
                                                             latf, latb, kl_sum);
    k3a_lse<<<NB / 128, 256, 0, stream>>>(latb, pb, pnorm, lse, idx);
    k3b_vq <<<NB / 128, 256, 0, stream>>>(latb, latf, pb, pnorm, protos, lse, idx,
                                          softacc, outq, diff_sum);
    k4_final<<<1, 256, 0, stream>>>(softacc, kl_sum, diff_sum, outq);
}
